// OptimizedLegalAttentionEngine_84456236908951
// MI455X (gfx1250) — compile-verified, matrix operand order checked by reading, not run on hardware
//
#include <hip/hip_runtime.h>
#include <hip/hip_bf16.h>

typedef __bf16 bf16;
typedef __attribute__((ext_vector_type(16))) __bf16 v16bf;
typedef __attribute__((ext_vector_type(8)))  float  v8f;
typedef long long ll;

#define cL  4
#define cA  4
#define cD  1024
#define cDK 256
#define cS  3072
#define cD2 2048

// ---------------- CDNA5 async global->LDS copy (ASYNCcnt path) ----------------
__device__ __forceinline__ void async_b128(void* lds, const void* g) {
  // low 32 bits of the generic pointer to a __shared__ object are the LDS offset
  unsigned l = (unsigned)(unsigned long long)lds;
  asm volatile("global_load_async_to_lds_b128 %0, %1, off"
               :: "v"(l), "v"(g) : "memory");
}
__device__ __forceinline__ void wait_async0() {
  asm volatile("s_wait_asynccnt 0x0" ::: "memory");
}
__device__ __forceinline__ void wait_async8() {
  asm volatile("s_wait_asynccnt 0x8" ::: "memory");
}

// ---------------- elementwise converts ----------------
__global__ void k_copy_convert(const float* __restrict__ src, float* __restrict__ dstF,
                               bf16* __restrict__ dstB, ll n) {
  ll i = (ll)blockIdx.x * blockDim.x + threadIdx.x;
  if (i < n) { float v = src[i]; dstF[i] = v; dstB[i] = (bf16)v; }
}

// transpose-convert: src [K][N] f32 -> dst [N][K] bf16 (K,N multiples of 32)
__launch_bounds__(256)
__global__ void k_f32_to_bf16_t(const float* __restrict__ src, bf16* __restrict__ dst,
                                int K, int N) {
  __shared__ float t[32][33];
  int k0 = blockIdx.x * 32, n0 = blockIdx.y * 32;
  int tx = threadIdx.x & 31, ty = threadIdx.x >> 5;   // 8 rows per pass
  #pragma unroll
  for (int r = ty; r < 32; r += 8)
    t[r][tx] = src[(ll)(k0 + r) * N + n0 + tx];
  __syncthreads();
  #pragma unroll
  for (int r = ty; r < 32; r += 8)
    dst[(ll)(n0 + r) * K + k0 + tx] = (bf16)t[tx][r];
}

// ---------------- bf16 WMMA GEMM, double-buffered async staging ----------------
// C[M,N] = alpha * A[M,K] x B[N,K]^T + bias ; optional ReLU
// A row-major (lda), B row-major-along-K (ldb)  -> both K-contiguous.
// outF fp32 / outB bf16 with leading dim ldc; tout=1 writes outB[col*ldc+row].
// REQUIRES: M,N multiples of 128; K multiple of 64; 16B-aligned pointers.
__launch_bounds__(256)
__global__ void k_gemm_bf16(const bf16* __restrict__ Ag, int lda,
                            const bf16* __restrict__ Bg, int ldb,
                            const float* __restrict__ bias,
                            float* __restrict__ outF, bf16* __restrict__ outB, int ldc,
                            int tout, int M, int N, int K, float alpha, int relu)
{
  // K-step 64; row stride 72 elems (144B) keeps 16B chunk alignment
  __shared__ __attribute__((aligned(16))) bf16 As[2][128][72];  // [buf][m][k]
  __shared__ __attribute__((aligned(16))) bf16 Bs[2][128][72];  // [buf][n][k]

  const int tid  = threadIdx.x;
  const int lane = tid & 31;
  const int wave = tid >> 5;
  const int hi   = lane >> 4;          // wave32 half
  const int l15  = lane & 15;
  const int wm   = (wave & 3) * 32;    // wave row offset in tile
  const int wn   = (wave >> 2) * 64;   // wave col offset in tile
  const int bm   = blockIdx.x * 128;
  const int bn   = blockIdx.y * 128;

  // staging: each thread owns 16B chunks at rows ra+{0,32,64,96}, k-chunk ca
  const int ra = tid >> 3;
  const int ca = (tid & 7) * 8;
  const bf16* Abase = Ag + (ll)(bm + ra) * lda + ca;
  const bf16* Bbase = Bg + (ll)(bn + ra) * ldb + ca;
  const ll a32 = (ll)32 * lda, b32 = (ll)32 * ldb;

  v8f acc[2][4];
  #pragma unroll
  for (int i = 0; i < 2; ++i)
    #pragma unroll
    for (int j = 0; j < 4; ++j)
      #pragma unroll
      for (int e = 0; e < 8; ++e) acc[i][j][e] = 0.0f;

  auto issue = [&](int buf, int k) {
    const bf16* Ap = Abase + k;
    const bf16* Bp = Bbase + k;
    async_b128(&As[buf][ra     ][ca], Ap);
    async_b128(&As[buf][ra + 32][ca], Ap + a32);
    async_b128(&As[buf][ra + 64][ca], Ap + 2 * a32);
    async_b128(&As[buf][ra + 96][ca], Ap + 3 * a32);
    async_b128(&Bs[buf][ra     ][ca], Bp);
    async_b128(&Bs[buf][ra + 32][ca], Bp + b32);
    async_b128(&Bs[buf][ra + 64][ca], Bp + 2 * b32);
    async_b128(&Bs[buf][ra + 96][ca], Bp + 3 * b32);
  };

  auto compute = [&](int buf) {
    // two 32-deep WMMA passes per staged 64-deep tile
    #pragma unroll
    for (int ks = 0; ks < 64; ks += 32) {
      // fragment gather per CDNA5 16-bit A/B layout: e -> K = ks+((e<8)?e:e+8)+8*half
      v16bf afrag[2], bfrag[4];
      #pragma unroll
      for (int i = 0; i < 2; ++i) {
        int row = wm + i * 16 + l15;
        #pragma unroll
        for (int e = 0; e < 16; ++e) {
          int kk = ks + ((e < 8) ? e : e + 8) + 8 * hi;
          afrag[i][e] = As[buf][row][kk];
        }
      }
      #pragma unroll
      for (int j = 0; j < 4; ++j) {
        int col = wn + j * 16 + l15;
        #pragma unroll
        for (int e = 0; e < 16; ++e) {
          int kk = ks + ((e < 8) ? e : e + 8) + 8 * hi;
          bfrag[j][e] = Bs[buf][col][kk];
        }
      }
      #pragma unroll
      for (int i = 0; i < 2; ++i)
        #pragma unroll
        for (int j = 0; j < 4; ++j)
          acc[i][j] = __builtin_amdgcn_wmma_f32_16x16x32_bf16(
              false, afrag[i], false, bfrag[j], (short)0, acc[i][j], false, false);
    }
  };

  // software pipeline: steady state always issues next tile + waits <=8
  issue(0, 0);
  int pb = 0;
  int k0 = 0;
  for (; k0 < K - 64; k0 += 64, pb ^= 1) {
    issue(pb ^ 1, k0 + 64);
    wait_async8();    // retire oldest 8 = current tile; next 8 stay in flight
    __syncthreads();  // all waves' copies for current tile visible
    compute(pb);
    __syncthreads();  // buffer pb fully consumed before re-issue
  }
  // peeled last tile
  wait_async0();
  __syncthreads();
  compute(pb);

  // epilogue: D layout -> row = base + e + 8*half, col = base + (lane&15)
  #pragma unroll
  for (int i = 0; i < 2; ++i) {
    #pragma unroll
    for (int j = 0; j < 4; ++j) {
      int col = bn + wn + j * 16 + l15;
      #pragma unroll
      for (int e = 0; e < 8; ++e) {
        int row = bm + wm + i * 16 + e + 8 * hi;
        float v = acc[i][j][e] * alpha;
        if (bias) v += bias[col];
        if (relu) v = v > 0.0f ? v : 0.0f;
        if (outF) outF[(ll)row * ldc + col] = v;
        if (outB) {
          ll o = tout ? ((ll)col * ldc + row) : ((ll)row * ldc + col);
          outB[o] = (bf16)v;
        }
      }
    }
  }
}

// ---------------- row softmax: fp32 scores -> bf16 probs ----------------
__launch_bounds__(256)
__global__ void k_softmax_bf16(const float* __restrict__ Sg, bf16* __restrict__ Pg, int ncol) {
  ll row = blockIdx.x;
  const float* x = Sg + row * (ll)ncol;
  bf16* p = Pg + row * (ll)ncol;
  __shared__ float red[256];
  int tid = threadIdx.x;
  float m = -3.4e38f;
  for (int c = tid; c < ncol; c += 256) m = fmaxf(m, x[c]);
  red[tid] = m; __syncthreads();
  for (int s = 128; s > 0; s >>= 1) { if (tid < s) red[tid] = fmaxf(red[tid], red[tid + s]); __syncthreads(); }
  m = red[0]; __syncthreads();
  float sum = 0.0f;
  for (int c = tid; c < ncol; c += 256) sum += __expf(x[c] - m);
  red[tid] = sum; __syncthreads();
  for (int s = 128; s > 0; s >>= 1) { if (tid < s) red[tid] += red[tid + s]; __syncthreads(); }
  float inv = 1.0f / red[0];
  for (int c = tid; c < ncol; c += 256) p[c] = (bf16)(__expf(x[c] - m) * inv);
}

// ---------------- fused residual add + LayerNorm ----------------
__launch_bounds__(256)
__global__ void k_add_ln(const float* __restrict__ x, const float* __restrict__ r,
                         const float* __restrict__ g, const float* __restrict__ b,
                         float* __restrict__ outF, bf16* __restrict__ outB, int n) {
  ll row = blockIdx.x;
  const float* xr = x + row * (ll)n;
  const float* rr = r + row * (ll)n;
  __shared__ float s1[256], s2[256];
  int tid = threadIdx.x;
  float a = 0.0f, q = 0.0f;
  for (int c = tid; c < n; c += 256) { float v = xr[c] + rr[c]; a += v; q += v * v; }
  s1[tid] = a; s2[tid] = q; __syncthreads();
  for (int s = 128; s > 0; s >>= 1) {
    if (tid < s) { s1[tid] += s1[tid + s]; s2[tid] += s2[tid + s]; }
    __syncthreads();
  }
  float mu = s1[0] / n;
  float var = s2[0] / n - mu * mu;
  float rs = rsqrtf(var + 1e-5f);
  for (int c = tid; c < n; c += 256) {
    float v = xr[c] + rr[c];
    float y = g[c] * (v - mu) * rs + b[c];
    outF[row * (ll)n + c] = y;
    outB[row * (ll)n + c] = (bf16)y;
  }
}

// ---------------- output heads (reads n_events/n_agents from device) ----------------
__launch_bounds__(256)
__global__ void k_heads(const float* __restrict__ h, const float* __restrict__ W,
                        const float* __restrict__ bh_, const int* __restrict__ dne,
                        const int* __restrict__ dna, float* __restrict__ out, int total) {
  int idx = blockIdx.x;
  int ne = *dne, na = *dna;
  if (idx >= total || idx >= 3 * na + ne) return;
  int head, row;
  if (idx < na)            { head = 0; row = ne + idx; }
  else if (idx < 2 * na)   { head = 1; row = ne + (idx - na); }
  else if (idx < 3 * na)   { head = 2; row = ne + (idx - 2 * na); }
  else                     { head = 3; row = idx - 3 * na; }
  if (row >= cS) return;
  __shared__ float red[256];
  int tid = threadIdx.x;
  float s = 0.0f;
  for (int c = tid; c < cD; c += 256) s += h[(ll)row * cD + c] * W[head * cD + c];
  red[tid] = s; __syncthreads();
  for (int st = 128; st > 0; st >>= 1) { if (tid < st) red[tid] += red[tid + st]; __syncthreads(); }
  if (tid == 0) out[idx] = red[0] + bh_[head];
}

// ---------------- host orchestration ----------------
extern "C" void kernel_launch(void* const* d_in, const int* in_sizes, int n_in,
                              void* d_out, int out_size, void* d_ws, size_t ws_size,
                              hipStream_t stream) {
  (void)in_sizes; (void)n_in; (void)ws_size;
  const float* hidden = (const float*)d_in[0];
  const float* Wq  = (const float*)d_in[1];  const float* bq  = (const float*)d_in[2];
  const float* Wk  = (const float*)d_in[3];  const float* bk  = (const float*)d_in[4];
  const float* Wv  = (const float*)d_in[5];  const float* bv  = (const float*)d_in[6];
  const float* Wh  = (const float*)d_in[7];  const float* bh  = (const float*)d_in[8];
  const float* Wo  = (const float*)d_in[9];  const float* bo  = (const float*)d_in[10];
  const float* g1  = (const float*)d_in[11]; const float* b1n = (const float*)d_in[12];
  const float* g2  = (const float*)d_in[13]; const float* b2n = (const float*)d_in[14];
  const float* Wf1 = (const float*)d_in[15]; const float* bf1 = (const float*)d_in[16];
  const float* Wf2 = (const float*)d_in[17]; const float* bf2 = (const float*)d_in[18];
  const float* Whead = (const float*)d_in[19]; const float* bhead = (const float*)d_in[20];
  const int* dne = (const int*)d_in[21];
  const int* dna = (const int*)d_in[22];
  float* out = (float*)d_out;

  char* ws = (char*)d_ws;
  size_t off = 0;
  auto alloc = [&](size_t bytes) -> void* {
    off = (off + 255) & ~(size_t)255;
    void* p = ws + off; off += bytes; return p;
  };

  // all weights stored TRANSPOSED in bf16: B operand is [N][K], K-contiguous
  bf16* wq_t  = (bf16*)alloc((size_t)cL * cD * cD  * 2);
  bf16* wk_t  = (bf16*)alloc((size_t)cL * cD * cD  * 2);
  bf16* wv_t  = (bf16*)alloc((size_t)cL * cD * cD  * 2);
  bf16* wh_t  = (bf16*)alloc((size_t)cL * cA * cDK * cD * 2);   // [l][a][DK][D]
  bf16* wo_t  = (bf16*)alloc((size_t)cL * cD * cD  * 2);
  bf16* wf1_t = (bf16*)alloc((size_t)cL * cD2 * cD * 2);        // [l][2D][D]
  bf16* wf2_t = (bf16*)alloc((size_t)cL * cD * cD2 * 2);        // [l][D][2D]
  float* hF    = (float*)alloc((size_t)cS * cD * 4);
  bf16*  h_bf  = (bf16*) alloc((size_t)cS * cD * 2);
  bf16*  q_bf  = (bf16*) alloc((size_t)cS * cD * 2);
  bf16*  kk_bf = (bf16*) alloc((size_t)cS * cD * 2);
  bf16*  vv_bf = (bf16*) alloc((size_t)cS * cD * 2);
  bf16*  qh_bf = (bf16*) alloc((size_t)cA * cS * cDK * 2);      // [a][S][DK]
  bf16*  kh_bf = (bf16*) alloc((size_t)cA * cS * cDK * 2);      // [a][S][DK]
  bf16*  vh_t  = (bf16*) alloc((size_t)cA * cDK * cS * 2);      // [a][DK][S] (transposed)
  float* scoresF = (float*)alloc((size_t)cA * cS * cS * 4);
  bf16*  attn_bf = (bf16*) alloc((size_t)cA * cS * cS * 2);
  bf16*  ctx_bf  = (bf16*) alloc((size_t)cS * cD * 2);
  float* tmpF    = (float*)alloc((size_t)cS * cD * 4);
  bf16*  ffn1_bf = (bf16*) alloc((size_t)cS * cD2 * 2);

  auto cvt_t = [&](const float* s, bf16* d, int K, int N) {
    k_f32_to_bf16_t<<<dim3(K / 32, N / 32), 256, 0, stream>>>(s, d, K, N);
  };
  for (int l = 0; l < cL; ++l) {
    cvt_t(Wq + (ll)l * cD * cD,  wq_t + (ll)l * cD * cD,  cD, cD);
    cvt_t(Wk + (ll)l * cD * cD,  wk_t + (ll)l * cD * cD,  cD, cD);
    cvt_t(Wv + (ll)l * cD * cD,  wv_t + (ll)l * cD * cD,  cD, cD);
    cvt_t(Wo + (ll)l * cD * cD,  wo_t + (ll)l * cD * cD,  cD, cD);
    cvt_t(Wf1 + (ll)l * cD * cD2, wf1_t + (ll)l * cD2 * cD, cD, cD2);
    cvt_t(Wf2 + (ll)l * cD2 * cD, wf2_t + (ll)l * cD * cD2, cD2, cD);
    for (int a = 0; a < cA; ++a)
      cvt_t(Wh + ((ll)l * cA + a) * cD * cDK, wh_t + ((ll)l * cA + a) * cDK * cD, cD, cDK);
  }

  k_copy_convert<<<(unsigned)(((ll)cS * cD + 255) / 256), 256, 0, stream>>>(
      hidden, hF, h_bf, (ll)cS * cD);

  auto gemm = [&](const bf16* Ag, int lda, const bf16* Bg, int ldb,
                  const float* bias, float* oF, bf16* oB, int ldc, int tout,
                  int M, int N, int K, float alpha, int relu) {
    dim3 grid(M / 128, N / 128);
    k_gemm_bf16<<<grid, 256, 0, stream>>>(Ag, lda, Bg, ldb, bias, oF, oB, ldc, tout,
                                          M, N, K, alpha, relu);
  };

  const float scale = 1.0f / 16.0f;  // 1/sqrt(DK=256)

  for (int l = 0; l < cL; ++l) {
    // QKV projections (bf16 outputs)
    gemm(h_bf, cD, wq_t + (ll)l * cD * cD, cD, bq + l * cD, nullptr, q_bf,  cD, 0, cS, cD, cD, 1.0f, 0);
    gemm(h_bf, cD, wk_t + (ll)l * cD * cD, cD, bk + l * cD, nullptr, kk_bf, cD, 0, cS, cD, cD, 1.0f, 0);
    gemm(h_bf, cD, wv_t + (ll)l * cD * cD, cD, bv + l * cD, nullptr, vv_bf, cD, 0, cS, cD, cD, 1.0f, 0);

    // per-head projections; Vh written transposed [DK][S] so it is K-contiguous as B
    for (int a = 0; a < cA; ++a) {
      const bf16* w = wh_t + ((ll)l * cA + a) * cDK * cD;
      const float* bha = bh + ((ll)l * cA + a) * cDK;
      gemm(q_bf,  cD, w, cD, bha, nullptr, qh_bf + (ll)a * cS * cDK, cDK, 0, cS, cDK, cD, 1.0f, 0);
      gemm(kk_bf, cD, w, cD, bha, nullptr, kh_bf + (ll)a * cS * cDK, cDK, 0, cS, cDK, cD, 1.0f, 0);
      gemm(vv_bf, cD, w, cD, bha, nullptr, vh_t  + (ll)a * cDK * cS, cS,  1, cS, cDK, cD, 1.0f, 0);
    }

    // scores = scale * Qh @ Kh^T   (Kh [S][DK] is already [N][K] form)
    for (int a = 0; a < cA; ++a)
      gemm(qh_bf + (ll)a * cS * cDK, cDK, kh_bf + (ll)a * cS * cDK, cDK,
           nullptr, scoresF + (ll)a * cS * cS, nullptr, cS, 0, cS, cS, cDK, scale, 0);

    // softmax rows -> bf16 probs
    k_softmax_bf16<<<cA * cS, 256, 0, stream>>>(scoresF, attn_bf, cS);

    // ctx = attn @ Vh  (Vh_t [DK][S] is [N][K] form); write interleaved into [S, D]
    for (int a = 0; a < cA; ++a)
      gemm(attn_bf + (ll)a * cS * cS, cS, vh_t + (ll)a * cDK * cS, cS,
           nullptr, nullptr, ctx_bf + a * cDK, cD, 0, cS, cDK, cS, 1.0f, 0);

    // attn_out = ctx @ Wo + bo (fp32); h = LN(h + attn_out)
    gemm(ctx_bf, cD, wo_t + (ll)l * cD * cD, cD, bo + l * cD, tmpF, nullptr, cD, 0, cS, cD, cD, 1.0f, 0);
    k_add_ln<<<cS, 256, 0, stream>>>(hF, tmpF, g1 + l * cD, b1n + l * cD, hF, h_bf, cD);

    // FFN
    gemm(h_bf, cD, wf1_t + (ll)l * cD2 * cD, cD, bf1 + l * cD2, nullptr, ffn1_bf, cD2, 0, cS, cD2, cD, 1.0f, 1);
    gemm(ffn1_bf, cD2, wf2_t + (ll)l * cD * cD2, cD2, bf2 + l * cD, tmpF, nullptr, cD, 0, cS, cD, cD2, 1.0f, 0);
    k_add_ln<<<cS, 256, 0, stream>>>(hF, tmpF, g2 + l * cD, b2n + l * cD, hF, h_bf, cD);
  }

  k_heads<<<out_size, 256, 0, stream>>>(hF, Whead, bhead, dne, dna, out, out_size);
}
